// GRUClassification_2637109920217
// MI455X (gfx1250) — compile-verified, matrix operand order checked by reading, not run on hardware
//
#include <hip/hip_runtime.h>
#include <hip/hip_bf16.h>

typedef __attribute__((ext_vector_type(16))) __bf16 v16bf;
typedef __attribute__((ext_vector_type(8)))  float  v8f;
typedef __attribute__((ext_vector_type(4)))  int    v4i;
typedef __attribute__((address_space(1))) v4i as1_v4i;   // global int4 vector
typedef __attribute__((address_space(3))) v4i as3_v4i;   // LDS int4 vector

constexpr int kB  = 1024;   // batch
constexpr int kT  = 80;     // timesteps
constexpr int kE  = 100;    // embedding dim
constexpr int kU  = 1024;   // GRU units
constexpr int kG  = 3072;   // 3*U gate width
constexpr int kKx1 = 128;   // E padded to multiple of 32
constexpr int kNT  = kG / 16; // 192 N-tiles of 16

// CDNA5 async global->LDS path (ASYNCcnt-tracked), with safe fallback.
#if __has_builtin(__builtin_amdgcn_global_load_async_to_lds_b128)
#  define HAVE_ASYNC_LDS 1
#else
#  define HAVE_ASYNC_LDS 0
#endif
#if HAVE_ASYNC_LDS
#  if __has_builtin(__builtin_amdgcn_s_wait_asynccnt)
#    define WAIT_STAGE(n) __builtin_amdgcn_s_wait_asynccnt(n)
#  else
#    define WAIT_STAGE(n) asm volatile("s_wait_asynccnt %0" ::"n"(n) : "memory")
#  endif
#else
#  define WAIT_STAGE(n) ((void)0)
#endif

// ---------------------------------------------------------------------------
// Pack a row-major (K x 3072) f32 weight matrix into per-lane bf16 WMMA
// B-fragment layout: P[((kk*192 + nt)*32 + lane)*16 + j] = W[k, n]
//   k = kk*32 + (lane>>4)*16 + j ; n = nt*16 + (lane&15)
// Rows k >= Ksrc are zero padding (for E=100 -> 128).
// ---------------------------------------------------------------------------
__global__ void pack_w_kernel(const float* __restrict__ W, __bf16* __restrict__ P,
                              int Ksrc, int Kpad) {
  int idx = blockIdx.x * blockDim.x + threadIdx.x;
  int total = Kpad * kG;
  if (idx >= total) return;
  int j    = idx & 15;
  int lane = (idx >> 4) & 31;
  int nt   = (idx >> 9) % kNT;
  int kk   = idx / (kNT << 9);
  int k = kk * 32 + ((lane >> 4) << 4) + j;
  int n = (nt << 4) + (lane & 15);
  float v = (k < Ksrc) ? W[(size_t)k * kG + n] : 0.f;
  P[idx] = (__bf16)v;
}

// ---------------------------------------------------------------------------
// Embedding gather -> bf16, layout [t][b][128] (E padded 100->128 with zeros)
// ---------------------------------------------------------------------------
__global__ void embed_kernel(const int* __restrict__ tokens,
                             const float* __restrict__ emb,
                             __bf16* __restrict__ x) {
  int idx = blockIdx.x * blockDim.x + threadIdx.x;
  if (idx >= kT * kB * 128) return;
  int e = idx & 127;
  int b = (idx >> 7) & (kB - 1);
  int t = idx >> 17;               // 128*1024 = 2^17
  int tok = tokens[b * kT + t];
  float v = (e < kE) ? emb[(size_t)tok * kE + e] : 0.f;
  x[idx] = (__bf16)v;
}

__global__ void zero_bf16_kernel(__bf16* __restrict__ p, int n) {
  int i = blockIdx.x * blockDim.x + threadIdx.x;
  if (i < n) p[i] = (__bf16)0.f;
}

__device__ __forceinline__ float sigmoidf_(float x) {
  return 1.f / (1.f + __expf(-x));
}

// ---------------------------------------------------------------------------
// Stage one K-tile of B fragments (3 gates x 4 n-subtiles = 12 KB) into LDS.
// 256 threads x 3 passes x 16 B. Uses CDNA5 async global->LDS when available.
// ---------------------------------------------------------------------------
__device__ __forceinline__ void stage_tile(const __bf16* __restrict__ src,
                                           int kk, int nb0, int4* dst, int tid) {
#pragma unroll
  for (int p = 0; p < 3; ++p) {
    int u = tid + p * 256;      // 0..767 16-byte units (12 KB)
    int g = u >> 8;             // gate chunk (4 KB each)
    int r = u & 255;            // 16B unit within chunk
    const __bf16* gp = src + (((size_t)(kk * kNT + g * 64 + nb0)) << 9) + r * 8;
#if HAVE_ASYNC_LDS
    __builtin_amdgcn_global_load_async_to_lds_b128((as1_v4i*)gp,
                                                   (as3_v4i*)&dst[u], 0, 0);
#else
    dst[u] = *(const int4*)gp;
#endif
  }
}

// 12 WMMAs against the staged LDS tile. Fragments live at f*1KB, f = g*4+s.
// B fragment loads are double-buffered so DScnt waits stay partial.
__device__ __forceinline__ void mma_tile(const __bf16* lds, v16bf a, int lane,
                                         v8f (&az)[4], v8f (&ar)[4],
                                         v8f (&ah)[4]) {
  const __bf16* p = lds + lane * 16;
  v16bf frag[2];
  frag[0] = *(const v16bf*)p;
#pragma unroll
  for (int f = 0; f < 12; ++f) {
    if (f < 11) frag[(f + 1) & 1] = *(const v16bf*)(p + (f + 1) * 512);
    v8f& acc = (f < 4) ? az[f] : (f < 8) ? ar[f - 4] : ah[f - 8];
    acc = __builtin_amdgcn_wmma_f32_16x16x32_bf16(
        false, a, false, frag[f & 1], (short)0, acc, false, false);
  }
}

// ---------------------------------------------------------------------------
// One fused GRU timestep (Keras reset_after semantics):
//   z = sig(xz+rz); r = sig(xr+rr); hh = tanh(xh + r*rh)
//   h_new = z*h + (1-z)*hh
// 256 threads (8 waves) compute a 128(row) x 64(col) tile of h_new. Each wave
// owns 16 rows x 64 cols = 4 n-subtiles x 4 accumulator sets (z, r, xh, rh).
// B fragments flow through a 4-deep LDS pipeline fed by async loads staged
// two K-tiles ahead; one barrier per K-tile.
// ---------------------------------------------------------------------------
__global__ __launch_bounds__(256) void gru_step_kernel(
    const __bf16* __restrict__ xt,     // (B, kx) row-major bf16
    int kx,                            // 128 (layer1) or 1024 (layer2)
    const __bf16* __restrict__ Wxp,    // packed (kx x 3072)
    const __bf16* __restrict__ Uhp,    // packed (1024 x 3072)
    const float* __restrict__ b_in,    // (3072,)
    const float* __restrict__ b_rec,   // (3072,)
    const __bf16* __restrict__ h_prev, // (B, 1024) bf16
    __bf16* __restrict__ h_new) {      // (B, 1024) bf16
  __shared__ alignas(32) int4 smem[4][768];   // 4 x 12 KB pipeline

  const int tid  = threadIdx.x;
  const int lane = tid & 31;
  const int wave = tid >> 5;
  const int n0   = blockIdx.x * 64;
  const int nb0  = n0 >> 4;
  const int row0 = blockIdx.y * 128 + wave * 16;
  const int nl   = lane & 15;
  const int kh   = lane >> 4;

  v8f acc_z[4], acc_r[4], acc_xh[4], acc_rh[4];
#pragma unroll
  for (int s = 0; s < 4; ++s) {
    int col = n0 + s * 16 + nl;
    float bz = b_in[col]          + b_rec[col];
    float br = b_in[kU + col]     + b_rec[kU + col];
    float bx = b_in[2 * kU + col];
    float bh = b_rec[2 * kU + col];
#pragma unroll
    for (int i = 0; i < 8; ++i) {
      acc_z[s][i]  = bz;
      acc_r[s][i]  = br;
      acc_xh[s][i] = bx;
      acc_rh[s][i] = bh;
    }
  }

  const int xtiles = kx >> 5;          // 4 or 32
  const int tot    = xtiles + (kU >> 5);
  const __bf16* xrow = xt     + (size_t)(row0 + nl) * kx + kh * 16;
  const __bf16* hrow = h_prev + (size_t)(row0 + nl) * kU + kh * 16;

  auto stage = [&](int i) {
    if (i < xtiles) stage_tile(Wxp, i,          nb0, smem[i & 3], tid);
    else            stage_tile(Uhp, i - xtiles, nb0, smem[i & 3], tid);
  };

  // Pipeline: stage(i+2) overlaps compute(i); one barrier per tile.
  // 3 async ops per stage -> wait<=6 guarantees stage(i) has landed.
  stage(0);
  stage(1);
  for (int i = 0; i < xtiles; ++i) {          // x-projection tiles
    stage(i + 2);
    WAIT_STAGE(6);
    __syncthreads();
    v16bf a = *(const v16bf*)(xrow + (size_t)i * 32);
    mma_tile((const __bf16*)smem[i & 3], a, lane, acc_z, acc_r, acc_xh);
  }
  for (int i = xtiles; i < tot - 2; ++i) {    // recurrent tiles (steady state)
    stage(i + 2);
    WAIT_STAGE(6);
    __syncthreads();
    v16bf a = *(const v16bf*)(hrow + (size_t)(i - xtiles) * 32);
    mma_tile((const __bf16*)smem[i & 3], a, lane, acc_z, acc_r, acc_rh);
  }
  {                                           // tail: two staged tiles remain
    int i = tot - 2;
    WAIT_STAGE(3);
    __syncthreads();
    v16bf a = *(const v16bf*)(hrow + (size_t)(i - xtiles) * 32);
    mma_tile((const __bf16*)smem[i & 3], a, lane, acc_z, acc_r, acc_rh);
  }
  {
    int i = tot - 1;
    WAIT_STAGE(0);
    __syncthreads();
    v16bf a = *(const v16bf*)(hrow + (size_t)(i - xtiles) * 32);
    mma_tile((const __bf16*)smem[i & 3], a, lane, acc_z, acc_r, acc_rh);
  }

  // ---- gate epilogue + state update ----
#pragma unroll
  for (int s = 0; s < 4; ++s) {
    int col = n0 + s * 16 + nl;
#pragma unroll
    for (int i = 0; i < 8; ++i) {
      int row = row0 + kh * 8 + i;
      float hp = (float)h_prev[(size_t)row * kU + col];
      float z  = sigmoidf_(acc_z[s][i]);
      float r  = sigmoidf_(acc_r[s][i]);
      float hh = tanhf(acc_xh[s][i] + r * acc_rh[s][i]);
      h_new[(size_t)row * kU + col] = (__bf16)(z * hp + (1.f - z) * hh);
    }
  }
}

// ---------------------------------------------------------------------------
// Dense head: out[b] = sigmoid(h2[b,:] . Wd + bd). One wave per row.
// ---------------------------------------------------------------------------
__global__ __launch_bounds__(256) void dense_kernel(
    const __bf16* __restrict__ h2, const float* __restrict__ Wd,
    const float* __restrict__ bd, float* __restrict__ out) {
  int row  = blockIdx.x * 8 + (threadIdx.x >> 5);
  int lane = threadIdx.x & 31;
  float acc = 0.f;
  for (int k = lane; k < kU; k += 32)
    acc += (float)h2[(size_t)row * kU + k] * Wd[k];
#pragma unroll
  for (int off = 16; off; off >>= 1) acc += __shfl_xor(acc, off, 32);
  if (lane == 0) out[row] = sigmoidf_(acc + bd[0]);
}

// ---------------------------------------------------------------------------
extern "C" void kernel_launch(void* const* d_in, const int* in_sizes, int n_in,
                              void* d_out, int out_size, void* d_ws, size_t ws_size,
                              hipStream_t stream) {
  const int*   tokens = (const int*)d_in[0];
  const float* emb    = (const float*)d_in[1];
  const float* W1     = (const float*)d_in[2];
  const float* U1     = (const float*)d_in[3];
  const float* b1     = (const float*)d_in[4];
  const float* W2     = (const float*)d_in[5];
  const float* U2     = (const float*)d_in[6];
  const float* b2     = (const float*)d_in[7];
  const float* Wd     = (const float*)d_in[8];
  const float* bd     = (const float*)d_in[9];
  float* out = (float*)d_out;

  char* ws = (char*)d_ws;
  size_t off = 0;
  auto walloc = [&](size_t bytes) -> void* {
    void* p = ws + off;
    off = (off + bytes + 255) & ~(size_t)255;
    return p;
  };
  __bf16* xemb = (__bf16*)walloc((size_t)kT * kB * 128 * 2); // 20 MB
  __bf16* W1p  = (__bf16*)walloc((size_t)kKx1 * kG * 2);
  __bf16* U1p  = (__bf16*)walloc((size_t)kU * kG * 2);
  __bf16* W2p  = (__bf16*)walloc((size_t)kU * kG * 2);
  __bf16* U2p  = (__bf16*)walloc((size_t)kU * kG * 2);
  __bf16* h1a  = (__bf16*)walloc((size_t)kB * kU * 2);
  __bf16* h1b  = (__bf16*)walloc((size_t)kB * kU * 2);
  __bf16* h2a  = (__bf16*)walloc((size_t)kB * kU * 2);
  __bf16* h2b  = (__bf16*)walloc((size_t)kB * kU * 2);

  // Phase 0: pack weights into WMMA B-fragment layout, gather embeddings,
  // zero the initial hidden states.
  pack_w_kernel<<<(kKx1 * kG + 255) / 256, 256, 0, stream>>>(W1, W1p, kE, kKx1);
  pack_w_kernel<<<(kU * kG + 255) / 256, 256, 0, stream>>>(U1, U1p, kU, kU);
  pack_w_kernel<<<(kU * kG + 255) / 256, 256, 0, stream>>>(W2, W2p, kU, kU);
  pack_w_kernel<<<(kU * kG + 255) / 256, 256, 0, stream>>>(U2, U2p, kU, kU);
  embed_kernel<<<(kT * kB * 128 + 255) / 256, 256, 0, stream>>>(tokens, emb, xemb);
  int hn = kB * kU;
  zero_bf16_kernel<<<(hn + 255) / 256, 256, 0, stream>>>(h1a, hn);
  zero_bf16_kernel<<<(hn + 255) / 256, 256, 0, stream>>>(h1b, hn);
  zero_bf16_kernel<<<(hn + 255) / 256, 256, 0, stream>>>(h2a, hn);
  zero_bf16_kernel<<<(hn + 255) / 256, 256, 0, stream>>>(h2b, hn);

  // Phase 1: 80 fused GRU timesteps per layer, pipelined so layer 2 of step t
  // consumes layer 1's freshly written state. Ping-pong h buffers.
  __bf16* h1[2] = {h1a, h1b};
  __bf16* h2[2] = {h2a, h2b};
  dim3 grid(kU / 64, kB / 128);   // 16 x 8 workgroups, 256 threads each
  for (int t = 0; t < kT; ++t) {
    int cur = t & 1, nxt = cur ^ 1;
    gru_step_kernel<<<grid, 256, 0, stream>>>(
        xemb + (size_t)t * kB * 128, kKx1, W1p, U1p,
        b1, b1 + kG, h1[cur], h1[nxt]);
    gru_step_kernel<<<grid, 256, 0, stream>>>(
        h1[nxt], kU, W2p, U2p,
        b2, b2 + kG, h2[cur], h2[nxt]);
  }

  // Phase 2: dense head on the final hidden state (ends in buffer 0 for T=80).
  dense_kernel<<<kB / 8, 256, 0, stream>>>(h2[kT & 1], Wd, bd, out);
}
